// GRUCell_7172595385053
// MI455X (gfx1250) — matrix operand order repaired and checked by reading, not compiled, not run on hardware
//
#include <hip/hip_runtime.h>
#include <hip/hip_bf16.h>

// ---------------------------------------------------------------------------
// Hyperbolic GRU cell for MI455X (gfx1250, wave32, WMMA).
// B=8192, H=512, D=4, HD=2048.
// 5 unique GEMMs [8192x2048]@[2048x2048] via V_WMMA_F32_16X16X32_BF16 with a
// bf16 hi/lo split (3 WMMAs per K=32 -> ~fp32 accuracy). Dual-B kernels share
// one A-staging between the (W_z,W_r) and (U_z,U_r) pairs. Ping-pong LDS
// double buffering hides global->LDS staging behind the matrix pipe.
// project_hyp_vec on the A operand is fused into staging (D=4 groups are
// contiguous along K).
// ---------------------------------------------------------------------------

#define BB   8192
#define HH   512
#define HD   2048           // K and N of every GEMM
#define EPSV 1e-15f
#define PROJ_EPS 1e-05f

typedef __bf16 bf16_t;
typedef __attribute__((ext_vector_type(16))) __bf16 v16bf;
typedef __attribute__((ext_vector_type(8)))  __bf16 v8bf;
typedef __attribute__((ext_vector_type(8)))  float  v8f;

union AFrag { v16bf v; v8bf h[2]; };

// ---------------- small per-4-group hyperbolic helpers ----------------------

__device__ __forceinline__ float dot4(const float* a, const float* b) {
    return a[0]*b[0] + a[1]*b[1] + a[2]*b[2] + a[3]*b[3];
}

__device__ __forceinline__ void proj4(const float* in, float* out) {
    float n2 = dot4(in, in);
    float s  = (1.0f - PROJ_EPS) / fmaxf(n2, 1.0f - PROJ_EPS);
    out[0] = in[0]*s; out[1] = in[1]*s; out[2] = in[2]*s; out[3] = in[3]*s;
}

__device__ __forceinline__ float norm4eps(const float* v) {  // ||v + EPS||
    float a = v[0]+EPSV, b = v[1]+EPSV, c = v[2]+EPSV, d = v[3]+EPSV;
    return sqrtf(a*a + b*b + c*c + d*d);
}

__device__ __forceinline__ float atanh_(float x) {
    return 0.5f * __logf((1.0f + x) / (1.0f - x));
}

__device__ __forceinline__ void mmd_post(const float* Mx, float xn, float* out) {
    float Mxn = norm4eps(Mx);
    float sc  = tanhf(Mxn / xn * atanh_(xn)) / Mxn;
    float t[4] = { Mx[0]*sc, Mx[1]*sc, Mx[2]*sc, Mx[3]*sc };
    proj4(t, out);
}

__device__ __forceinline__ void mob_add4(const float* u, const float* vin, float* out) {
    float v[4] = { vin[0]+EPSV, vin[1]+EPSV, vin[2]+EPSV, vin[3]+EPSV };
    float nuv = 2.0f * dot4(u, v);
    float nu  = dot4(u, u);
    float nv  = dot4(v, v);
    float denom = 1.0f + nuv + nv*nu;
    float cu = (1.0f + nuv + nv) / denom;
    float cv = (1.0f - nu) / denom;
    float t[4] = { cu*u[0]+cv*v[0], cu*u[1]+cv*v[1], cu*u[2]+cv*v[2], cu*u[3]+cv*v[3] };
    proj4(t, out);
}

__device__ __forceinline__ void logmap0_4(const float* y, float* out) {
    float t[4] = { y[0]+EPSV, y[1]+EPSV, y[2]+EPSV, y[3]+EPSV };
    float d[4]; proj4(t, d);
    float n  = sqrtf(dot4(d, d));
    float sc = atanh_(n) / n;
    out[0]=d[0]*sc; out[1]=d[1]*sc; out[2]=d[2]*sc; out[3]=d[3]*sc;
}

__device__ __forceinline__ void mob_pw4(const float* x, const float* u, float* out) {
    float t[4] = { x[0]+EPSV, x[1]+EPSV, x[2]+EPSV, x[3]+EPSV };
    float xx[4]; proj4(t, xx);
    float Mx[4] = { xx[0]*u[0], xx[1]*u[1], xx[2]*u[2], xx[3]*u[3] };
    float Mxn = norm4eps(Mx);
    float xn  = sqrtf(dot4(xx, xx));
    float sc  = tanhf(Mxn / xn * atanh_(xn)) / Mxn;
    float r[4] = { Mx[0]*sc, Mx[1]*sc, Mx[2]*sc, Mx[3]*sc };
    proj4(r, out);
}

__device__ __forceinline__ float sigmoidf(float x) {
    return 1.0f / (1.0f + __expf(-x));
}

__device__ __forceinline__ void split_bf16(float f, bf16_t& hi, bf16_t& lo) {
    hi = (bf16_t)f;
    lo = (bf16_t)(f - (float)hi);
}

// ---------------------------------------------------------------------------
// GEMM: C[m][M,N] = project_hyp_vec_groups4(A)[M,K] @ W[m][K,N], m < NMAT.
// M=8192, K=2048, per-launch N=2048. Block = 256 threads (8 waves),
// block tile 128M x 64N, wave tile 32x32 per matrix, ping-pong LDS.
// ---------------------------------------------------------------------------
template<int NMAT>
__global__ __launch_bounds__(256)
void gemm_proj_bf16x3(const float* __restrict__ A,
                      const float* __restrict__ W0, const float* __restrict__ W1,
                      float* __restrict__ C0, float* __restrict__ C1) {
    __shared__ __align__(32) bf16_t AhiS[2][128*32];
    __shared__ __align__(32) bf16_t AloS[2][128*32];
    __shared__ __align__(32) bf16_t BhiS[NMAT][2][64*32];   // [n][k], n-major
    __shared__ __align__(32) bf16_t BloS[NMAT][2][64*32];

    const int tid  = threadIdx.x;
    const int lane = tid & 31;
    const int wid  = tid >> 5;
    const int waveM = (wid & 3) * 32;   // 4 waves along M
    const int waveN = (wid >> 2) * 32;  // 2 waves along N
    const int blockN = blockIdx.x * 64;
    const int blockM = blockIdx.y * 128;
    const int lm = lane & 15;
    const int lh = lane >> 4;

    const float* Wm[2] = { W0, (NMAT == 2) ? W1 : W0 };
    float*       Cm[2] = { C0, (NMAT == 2) ? C1 : C0 };

    // ---- staging helpers (global fp32 -> proj -> bf16 hi/lo -> LDS) --------
    auto stage_tiles = [&](int kt, int buf) {
        // A tile: 128 rows x 32 K, projection fused per 4-group
        #pragma unroll
        for (int i = 0; i < 4; ++i) {
            int g   = tid + i*256;        // 0..1023
            int row = g >> 3;
            int kg  = (g & 7) * 4;
            const float4 v = *(const float4*)(A + (size_t)(blockM + row)*HD + kt + kg);
            float in[4] = { v.x, v.y, v.z, v.w };
            float pr[4]; proj4(in, pr);
            int o = row*32 + kg;
            #pragma unroll
            for (int j = 0; j < 4; ++j) {
                bf16_t h, l; split_bf16(pr[j], h, l);
                AhiS[buf][o + j] = h; AloS[buf][o + j] = l;
            }
        }
        // B tiles: 32 K x 64 N each, transposed into LDS [n][k]
        #pragma unroll
        for (int mm = 0; mm < NMAT; ++mm) {
            #pragma unroll
            for (int i = 0; i < 2; ++i) {
                int t  = tid + i*256;     // 0..511
                int n  = t & 63;          // consecutive tid -> consecutive n
                int kg = (t >> 6) * 4;
                int o  = n*32 + kg;
                #pragma unroll
                for (int j = 0; j < 4; ++j) {
                    float f = Wm[mm][(size_t)(kt + kg + j)*HD + blockN + n];
                    bf16_t h, l; split_bf16(f, h, l);
                    BhiS[mm][buf][o + j] = h; BloS[mm][buf][o + j] = l;
                }
            }
        }
    };

    v8f acc[NMAT][2][2];
    #pragma unroll
    for (int mm = 0; mm < NMAT; ++mm)
        #pragma unroll
        for (int i = 0; i < 2; ++i)
            #pragma unroll
            for (int j = 0; j < 2; ++j)
                acc[mm][i][j] = (v8f)(0.0f);

    stage_tiles(0, 0);
    __syncthreads();

    int cur = 0;
    for (int kt = 0; kt < HD; kt += 32) {
        // ---- fragments from buf[cur] (same contiguous k order in A and B) ------------------
        // A 16x32: lane half 0 -> K 0..15, half 1 -> K 16..31 (contiguous)
        AFrag aH[2], aL[2];
        #pragma unroll
        for (int mt = 0; mt < 2; ++mt) {
            int m = waveM + mt*16 + lm;
            aH[mt].h[0] = *(const v8bf*)(AhiS[cur] + m*32     + lh*16);
            aH[mt].h[1] = *(const v8bf*)(AhiS[cur] + m*32 + 8 + lh*16);
            aL[mt].h[0] = *(const v8bf*)(AloS[cur] + m*32     + lh*16);
            aL[mt].h[1] = *(const v8bf*)(AloS[cur] + m*32 + 8 + lh*16);
        }
        // B 32x16: lane half 0 -> K 0..15, half 1 -> K 16..31 (contiguous)
        v16bf bH[NMAT][2], bL[NMAT][2];
        #pragma unroll
        for (int mm = 0; mm < NMAT; ++mm)
            #pragma unroll
            for (int nt = 0; nt < 2; ++nt) {
                int n = waveN + nt*16 + lm;
                bH[mm][nt] = *(const v16bf*)(BhiS[mm][cur] + n*32 + lh*16);
                bL[mm][nt] = *(const v16bf*)(BloS[mm][cur] + n*32 + lh*16);
            }

        // ---- stage next K-step into the other buffer while WMMAs run ------
        if (kt + 32 < HD) stage_tiles(kt + 32, cur ^ 1);
        // prefetch streaming A operand two steps ahead (weights sit in L2)
        if (kt + 64 < HD && tid < 128)
            __builtin_prefetch(A + (size_t)(blockM + tid)*HD + kt + 64, 0, 0);

        #pragma unroll
        for (int mm = 0; mm < NMAT; ++mm)
            #pragma unroll
            for (int mt = 0; mt < 2; ++mt)
                #pragma unroll
                for (int nt = 0; nt < 2; ++nt) {
                    v8f c = acc[mm][mt][nt];
                    c = __builtin_amdgcn_wmma_f32_16x16x32_bf16(
                            false, aH[mt].v, false, bH[mm][nt], (short)0, c, false, false);
                    c = __builtin_amdgcn_wmma_f32_16x16x32_bf16(
                            false, aH[mt].v, false, bL[mm][nt], (short)0, c, false, false);
                    c = __builtin_amdgcn_wmma_f32_16x16x32_bf16(
                            false, aL[mt].v, false, bH[mm][nt], (short)0, c, false, false);
                    acc[mm][mt][nt] = c;
                }
        __syncthreads();
        cur ^= 1;
    }

    // ---- store C: VGPR j -> M = j + 8*(lane>=16), N = lane&15 --------------
    #pragma unroll
    for (int mm = 0; mm < NMAT; ++mm)
        #pragma unroll
        for (int mt = 0; mt < 2; ++mt)
            #pragma unroll
            for (int nt = 0; nt < 2; ++nt) {
                v8f a = acc[mm][mt][nt];
                int col   = blockN + waveN + nt*16 + lm;
                int rbase = blockM + waveM + mt*16 + lh*8;
                #pragma unroll
                for (int j = 0; j < 8; ++j)
                    Cm[mm][(size_t)(rbase + j)*HD + col] = a[j];
            }
}

// ---------------------------------------------------------------------------
// Stage A (elementwise, 1 thread per (b,h) 4-group):
//  z = sigmoid(logmap0(mob_add(mob_add(mmd(Wh_z), mmd(Ux_z)), b_z)))  (same for r)
//  r_point_h = mob_pointwise_prod(hidden, r)
//  writes: z -> s2 (over Ux_z), r_point_h -> s0 (over Wh_z)
// ---------------------------------------------------------------------------
__global__ __launch_bounds__(256)
void stageA(const float* __restrict__ hyp_x, const float* __restrict__ hidden,
            float* __restrict__ s0, const float* __restrict__ s1,
            float* __restrict__ s2, const float* __restrict__ s3,
            const float* __restrict__ b_z, const float* __restrict__ b_r) {
    size_t g   = (size_t)blockIdx.x * 256 + threadIdx.x;   // < B*H
    size_t off = g * 4;
    int h_idx  = (int)(g & (HH - 1));

    float4 hv = *(const float4*)(hidden + off);
    float4 xv = *(const float4*)(hyp_x + off);
    float h4[4] = { hv.x, hv.y, hv.z, hv.w };
    float x4[4] = { xv.x, xv.y, xv.z, xv.w };

    float hp[4]; proj4(h4, hp); float hn = norm4eps(hp);
    float xp[4]; proj4(x4, xp); float xn = norm4eps(xp);

    float4 c0 = *(const float4*)(s0 + off);
    float4 c1 = *(const float4*)(s1 + off);
    float4 c2 = *(const float4*)(s2 + off);
    float4 c3 = *(const float4*)(s3 + off);

    // ---- z gate ----
    {
        float Mh[4] = { c0.x, c0.y, c0.z, c0.w };
        float Mx[4] = { c2.x, c2.y, c2.z, c2.w };
        float a[4], b[4]; mmd_post(Mh, hn, a); mmd_post(Mx, xn, b);
        float t1[4]; mob_add4(a, b, t1);
        float bz[4] = { b_z[h_idx*4+0], b_z[h_idx*4+1], b_z[h_idx*4+2], b_z[h_idx*4+3] };
        float tz[4]; mob_add4(t1, bz, tz);
        float lz[4]; logmap0_4(tz, lz);
        *(float4*)(s2 + off) = make_float4(sigmoidf(lz[0]), sigmoidf(lz[1]),
                                           sigmoidf(lz[2]), sigmoidf(lz[3]));
    }
    // ---- r gate + r_point_h ----
    {
        float Mh[4] = { c1.x, c1.y, c1.z, c1.w };
        float Mx[4] = { c3.x, c3.y, c3.z, c3.w };
        float a[4], b[4]; mmd_post(Mh, hn, a); mmd_post(Mx, xn, b);
        float t1[4]; mob_add4(a, b, t1);
        float br[4] = { b_r[h_idx*4+0], b_r[h_idx*4+1], b_r[h_idx*4+2], b_r[h_idx*4+3] };
        float tr[4]; mob_add4(t1, br, tr);
        float lr[4]; logmap0_4(tr, lr);
        float r4[4] = { sigmoidf(lr[0]), sigmoidf(lr[1]), sigmoidf(lr[2]), sigmoidf(lr[3]) };
        float rph[4]; mob_pw4(h4, r4, rph);
        *(float4*)(s0 + off) = make_float4(rph[0], rph[1], rph[2], rph[3]);
    }
}

// ---------------------------------------------------------------------------
// Stage B: h_tilde = mob_add(mob_add(mmd(W_h@rph), mmd(Ux_r)), b_h)
//          new_h   = mob_add(hidden, mob_pw(mob_add(-hidden, h_tilde), z))
// s0 = r_point_h, s1 = W_h @ proj(r_point_h), s2 = z, s3 = Ux_r
// ---------------------------------------------------------------------------
__global__ __launch_bounds__(256)
void stageB(const float* __restrict__ hyp_x, const float* __restrict__ hidden,
            const float* __restrict__ s0, const float* __restrict__ s1,
            const float* __restrict__ s2, const float* __restrict__ s3,
            const float* __restrict__ b_h, float* __restrict__ out) {
    size_t g   = (size_t)blockIdx.x * 256 + threadIdx.x;
    size_t off = g * 4;
    int h_idx  = (int)(g & (HH - 1));

    float4 hv = *(const float4*)(hidden + off);
    float4 xv = *(const float4*)(hyp_x + off);
    float4 rv = *(const float4*)(s0 + off);
    float4 wv = *(const float4*)(s1 + off);
    float4 zv = *(const float4*)(s2 + off);
    float4 uv = *(const float4*)(s3 + off);
    float h4[4]  = { hv.x, hv.y, hv.z, hv.w };
    float x4[4]  = { xv.x, xv.y, xv.z, xv.w };
    float rph[4] = { rv.x, rv.y, rv.z, rv.w };
    float Mh[4]  = { wv.x, wv.y, wv.z, wv.w };
    float z4[4]  = { zv.x, zv.y, zv.z, zv.w };
    float Mx[4]  = { uv.x, uv.y, uv.z, uv.w };

    float rp[4]; proj4(rph, rp); float rn = norm4eps(rp);
    float xp[4]; proj4(x4, xp);  float xn = norm4eps(xp);

    float a[4], b[4]; mmd_post(Mh, rn, a); mmd_post(Mx, xn, b);
    float t1[4]; mob_add4(a, b, t1);
    float bh[4] = { b_h[h_idx*4+0], b_h[h_idx*4+1], b_h[h_idx*4+2], b_h[h_idx*4+3] };
    float ht[4]; mob_add4(t1, bh, ht);

    float nh[4] = { -h4[0], -h4[1], -h4[2], -h4[3] };
    float mo[4]; mob_add4(nh, ht, mo);
    float pp[4]; mob_pw4(mo, z4, pp);
    float o4[4]; mob_add4(h4, pp, o4);
    *(float4*)(out + off) = make_float4(o4[0], o4[1], o4[2], o4[3]);
}

// ---------------------------------------------------------------------------
extern "C" void kernel_launch(void* const* d_in, const int* in_sizes, int n_in,
                              void* d_out, int out_size, void* d_ws, size_t ws_size,
                              hipStream_t stream) {
    const float* hyp_x  = (const float*)d_in[0];
    const float* hidden = (const float*)d_in[1];
    const float* W_z    = (const float*)d_in[2];
    const float* W_r    = (const float*)d_in[3];
    const float* W_h    = (const float*)d_in[4];
    const float* U_z    = (const float*)d_in[5];
    const float* U_r    = (const float*)d_in[6];
    // d_in[7] = U_h: unused by the reference (it faithfully reuses U_r)
    const float* b_z    = (const float*)d_in[8];
    const float* b_r    = (const float*)d_in[9];
    const float* b_h    = (const float*)d_in[10];
    float* out = (float*)d_out;

    const size_t slot = (size_t)BB * HD;       // 16M floats per slot
    float* s0 = (float*)d_ws;
    float* s1 = s0 + slot;
    float* s2 = s1 + slot;
    float* s3 = s2 + slot;

    dim3 gg(HD/64, BB/128);                    // 32 x 64 blocks
    dim3 bb(256);
    const int ngroups = (BB * HH) / 256;       // 16384 blocks for elementwise

    gemm_proj_bf16x3<2><<<gg, bb, 0, stream>>>(hidden, W_z, W_r, s0, s1);
    gemm_proj_bf16x3<2><<<gg, bb, 0, stream>>>(hyp_x,  U_z, U_r, s2, s3);
    stageA<<<ngroups, bb, 0, stream>>>(hyp_x, hidden, s0, s1, s2, s3, b_z, b_r);
    gemm_proj_bf16x3<1><<<gg, bb, 0, stream>>>(s0, W_h, nullptr, s1, nullptr);
    stageB<<<ngroups, bb, 0, stream>>>(hyp_x, hidden, s0, s1, s2, s3, b_h, out);
}
